// SpinnTreeLSTM_34110630265348
// MI455X (gfx1250) — matrix operand order repaired and checked by reading, not compiled, not run on hardware
//
#include <hip/hip_runtime.h>
#include <hip/hip_bf16.h>

// ---------------------------------------------------------------------------
// SPINN TreeLSTM on gfx1250 (MI455X): bf16 WMMA 16x16x32 with f32 accumulate.
// B=128, L=128, E=256, H=256, T=2L-1=255. SHIFT=1, REDUCE=2.
// ---------------------------------------------------------------------------

typedef __attribute__((ext_vector_type(16))) __bf16        v16bf;
typedef __attribute__((ext_vector_type(8)))  float         v8f;
typedef __attribute__((ext_vector_type(4)))  unsigned int  v4u;

union FragBF {
    v16bf v;
    unsigned short s[16];
    v4u q[2];           // two 16-byte halves -> b128 loads
};

__device__ __forceinline__ unsigned short f2bf(float f) {
    // round-to-nearest-even fp32 -> bf16
    unsigned int u = __float_as_uint(f);
    unsigned int r = (u + 0x7FFFu + ((u >> 16) & 1u)) >> 16;
    return (unsigned short)r;
}

__device__ __forceinline__ float sigf(float x) { return 1.0f / (1.0f + expf(-x)); }

#define B_   128
#define L_   128
#define E_   256
#define H_   256
#define H2_  512   // 2*H
#define G4_  1024  // 4*H

// A-fragment element K index for 16-bit 16x32 A layout (ISA 7.12.2):
//  lane half selects K 0-7 / 8-15 in elems 0-7 and K 16-23 / 24-31 in elems 8-15
__device__ __forceinline__ int a_elem_k(int half, int i) {
    return (i < 8) ? (8 * half + i) : (16 + 8 * half + (i - 8));
}

// ---------------------------------------------------------------------------
// Kernel 1: pre-swizzle weights into wave32 WMMA B-fragment layout (bf16).
//  Bfrag: [Wl;Wr] (512 x 1024) -> [ntile(64)][kstep(16)][lane(32)][elem(16)]
//  Cfrag/Ofrag: Wc/Wo (256 x 256) -> [ntile(16)][kstep(8)][lane(32)][elem(16)]
//  B-matrix layout: n = lane%16 ; k = kstep*32 + 16*(lane/16) + elem
// ---------------------------------------------------------------------------
__global__ __launch_bounds__(256) void prep_weights_kernel(
    const float* __restrict__ Wl, const float* __restrict__ Wr,
    const float* __restrict__ Wc, const float* __restrict__ Wo,
    unsigned short* __restrict__ Bfrag,
    unsigned short* __restrict__ Cfrag,
    unsigned short* __restrict__ Ofrag)
{
    int id = blockIdx.x * 256 + threadIdx.x;
    if (id < 524288) {                       // Bfrag: 64*16*32*16
        int i    = id & 15;
        int lane = (id >> 4) & 31;
        int ks   = (id >> 9) & 15;
        int nt   = id >> 13;                 // 0..63
        int n = nt * 16 + (lane & 15);
        int k = ks * 32 + ((lane >> 4) << 4) + i;
        float v = (k < H_) ? Wl[k * G4_ + n] : Wr[(k - H_) * G4_ + n];
        Bfrag[id] = f2bf(v);
    } else if (id < 524288 + 65536) {        // Cfrag: 16*8*32*16
        int id2  = id - 524288;
        int i    = id2 & 15;
        int lane = (id2 >> 4) & 31;
        int ks   = (id2 >> 9) & 7;
        int nt   = id2 >> 12;                // 0..15
        int n = nt * 16 + (lane & 15);
        int k = ks * 32 + ((lane >> 4) << 4) + i;
        Cfrag[id2] = f2bf(Wc[k * H_ + n]);
    } else if (id < 524288 + 131072) {       // Ofrag
        int id2  = id - 524288 - 65536;
        int i    = id2 & 15;
        int lane = (id2 >> 4) & 31;
        int ks   = (id2 >> 9) & 7;
        int nt   = id2 >> 12;
        int n = nt * 16 + (lane & 15);
        int k = ks * 32 + ((lane >> 4) << 4) + i;
        Ofrag[id2] = f2bf(Wo[k * H_ + n]);
    }
}

// ---------------------------------------------------------------------------
// Kernel 1b: pre-swizzle seq_embs (16384 x 256 rows of l*B+b) into bf16
// A-fragments: [mtile(1024)][kstep(8)][lane(32)][elem(16)].
// Pays the fp32->bf16 conversion once so the leaf GEMM loop is pure
// b128-load + WMMA.
// ---------------------------------------------------------------------------
__global__ __launch_bounds__(256) void prep_seq_kernel(
    const float* __restrict__ seq, unsigned short* __restrict__ Afrag)
{
    int id = blockIdx.x * 256 + threadIdx.x;     // 0 .. 4194303
    int i    = id & 15;
    int lane = (id >> 4) & 31;
    int ks   = (id >> 9) & 7;
    int mt   = id >> 12;                         // 0..1023
    int m = lane & 15, half = lane >> 4;
    int k = ks * 32 + 16 * half + i;
    Afrag[id] = f2bf(seq[(size_t)(mt * 16 + m) * E_ + k]);
}

// ---------------------------------------------------------------------------
// Kernel 2: sequential pointer scan (per batch element, trivial int ops).
// Precomputes all read/write stack slots for the T steps.
// ---------------------------------------------------------------------------
__global__ __launch_bounds__(128) void idx_kernel(
    const int* __restrict__ trans, int T,
    int* __restrict__ is_shift, int* __restrict__ liA, int* __restrict__ riA,
    int* __restrict__ biA, int* __restrict__ wrA, int* __restrict__ spf)
{
    int b = threadIdx.x;
    if (b >= B_) return;
    int sp = 0, bp = L_;
    for (int t = 0; t < T; ++t) {
        int tr = trans[t * B_ + b];
        int s  = (tr == 1);
        int o  = t * B_ + b;
        is_shift[o] = s;
        liA[o] = max(sp - 2, 0);
        riA[o] = max(sp - 1, 0);
        biA[o] = max(bp - 1, 0);
        wrA[o] = s ? sp : max(sp - 2, 0);
        sp = s ? sp + 1 : sp - 1;
        bp -= s;
    }
    spf[b] = max(sp - 1, 0);
}

// ---------------------------------------------------------------------------
// Kernel 3: leaf transform. GEMM (L*B=16384, E=256) x (E, H) for Wc and Wo
// sharing the A fragment, then h = sigmoid(o)*tanh(c); scatter into
// buffers[b][l][0:256]=h, [256:512]=c.
// One wave = one 16x16 output tile pair. 8 waves / block, 2048 blocks.
// Inner loop: 3x b128 fragment loads + 2x v_wmma per k-step.
// ---------------------------------------------------------------------------
__global__ __launch_bounds__(256) void leaf_kernel(
    const unsigned short* __restrict__ Afrag, const float* __restrict__ bc,
    const float* __restrict__ bo,
    const unsigned short* __restrict__ Cfrag,
    const unsigned short* __restrict__ Ofrag,
    float* __restrict__ buffers)
{
    int wave = threadIdx.x >> 5;
    int lane = threadIdx.x & 31;
    int tile = blockIdx.x * 8 + wave;        // 0..16383
    int mt = tile >> 4;                      // 0..1023  (M tile over L*B rows)
    int nt = tile & 15;                      // 0..15    (N tile over H cols)
    int half = lane >> 4;

    v8f accC = {}, accO = {};
    for (int ks = 0; ks < 8; ++ks) {
        FragBF a, bC, bO;
        const v4u* pa = (const v4u*)(Afrag + (((size_t)mt * 8 + ks) * 32 + lane) * 16);
        const v4u* pc = (const v4u*)(Cfrag + (((size_t)nt * 8 + ks) * 32 + lane) * 16);
        const v4u* po = (const v4u*)(Ofrag + (((size_t)nt * 8 + ks) * 32 + lane) * 16);
        a.q[0]  = pa[0]; a.q[1]  = pa[1];
        bC.q[0] = pc[0]; bC.q[1] = pc[1];
        bO.q[0] = po[0]; bO.q[1] = po[1];
        accC = __builtin_amdgcn_wmma_f32_16x16x32_bf16(false, a.v, false, bC.v,
                                                       (short)0, accC, false, false);
        accO = __builtin_amdgcn_wmma_f32_16x16x32_bf16(false, a.v, false, bO.v,
                                                       (short)0, accO, false, false);
    }

    int n = nt * 16 + (lane & 15);
    float bcv = bc[n], bov = bo[n];
    int r0 = mt * 16;
#pragma unroll
    for (int v = 0; v < 8; ++v) {
        int row = r0 + v + 8 * half;         // global row in (L*B)
        int l = row >> 7;                    // row / B
        int b = row & 127;                   // row % B
        float c = accC[v] + bcv;
        float o = accO[v] + bov;
        float h = sigf(o) * tanhf(c);
        float* bufb = buffers + ((size_t)b * L_ + l) * H2_;
        bufb[n]       = h;
        bufb[H_ + n]  = c;
    }
}

// ---------------------------------------------------------------------------
// Kernel 4: one scan step. Grid: 128 blocks (8 m-blocks x 16 j-blocks),
// 128 threads = 4 waves. Waves 0..3 compute gate chunks i/lf/rf/u for the
// same (16 batch rows, 16 j-cols) sharing a bf16 A tile staged in LDS.
// Writes new_state to staging[] (committed by a separate kernel to avoid
// cross-workgroup read/write races on the stack) and the pushed h output.
// ---------------------------------------------------------------------------
__global__ __launch_bounds__(128) void step_kernel(
    int t, int T,
    const float* __restrict__ stack, const float* __restrict__ buffers,
    const unsigned short* __restrict__ Bfrag, const float* __restrict__ bcomp,
    const int* __restrict__ is_shift, const int* __restrict__ liA,
    const int* __restrict__ riA, const int* __restrict__ biA,
    float* __restrict__ staging, float* __restrict__ out)
{
    __shared__ unsigned short Ash[16 * 512];     // 16 rows x (lh|rh) in bf16, 16KB
    __shared__ float gsh[4][16][16];

    int mblk = blockIdx.x >> 4;                  // 0..7
    int jblk = blockIdx.x & 15;                  // 0..15
    int tid  = threadIdx.x;
    int tB   = t * B_;

    // Phase 1: gather A = [lh | rh] for 16 batch rows into LDS (f32 -> bf16)
    for (int idx = tid; idx < 16 * 512; idx += 128) {
        int row = idx >> 9;
        int col = idx & 511;
        int b   = mblk * 16 + row;
        int src = (col < H_) ? liA[tB + b] : riA[tB + b];
        int cc  = (col < H_) ? col : (col - H_);
        float v = stack[((size_t)b * L_ + src) * H2_ + cc];
        Ash[idx] = f2bf(v);
    }
    __syncthreads();

    // Phase 2: WMMA. wave w handles gate chunk w -> global ntile = w*16 + jblk
    int wave = tid >> 5, lane = tid & 31;
    int m = lane & 15, half = lane >> 4;
    int ntg = wave * 16 + jblk;                  // 0..63
    v8f acc = {};
    for (int ks = 0; ks < 16; ++ks) {
        FragBF a, bb;
        // elems 0-15: K = ks*32 + 16*half + [0..15] -> 32B-aligned contiguous
        // two consecutive 16B chunks
        const v4u* pa = (const v4u*)&Ash[m * 512 + ks * 32 + 16 * half];
        a.q[0] = pa[0];
        a.q[1] = pa[1];
        const v4u* pb = (const v4u*)(Bfrag + (((size_t)ntg * 16 + ks) * 32 + lane) * 16);
        bb.q[0] = pb[0];
        bb.q[1] = pb[1];
        acc = __builtin_amdgcn_wmma_f32_16x16x32_bf16(false, a.v, false, bb.v,
                                                      (short)0, acc, false, false);
    }
#pragma unroll
    for (int v = 0; v < 8; ++v) gsh[wave][v + 8 * half][lane & 15] = acc[v];
    __syncthreads();

    // Phase 3: LSTM pointwise + shift/reduce select. thread = (row, jcol).
    int row = tid >> 4, jc = tid & 15;
    int b  = mblk * 16 + row;
    int jg = jblk * 16 + jc;                     // 0..255
    int s  = is_shift[tB + b];
    float h, c;
    if (s) {
        const float* bufb = buffers + ((size_t)b * L_ + biA[tB + b]) * H2_;
        h = bufb[jg];
        c = bufb[H_ + jg];
    } else {
        float gi = gsh[0][row][jc] + bcomp[jg];
        float gl = gsh[1][row][jc] + bcomp[H_ + jg];
        float gr = gsh[2][row][jc] + bcomp[2 * H_ + jg];
        float gu = gsh[3][row][jc] + bcomp[3 * H_ + jg];
        int li = liA[tB + b], ri = riA[tB + b];
        float lc = stack[((size_t)b * L_ + li) * H2_ + H_ + jg];
        float rc = stack[((size_t)b * L_ + ri) * H2_ + H_ + jg];
        c = sigf(gi) * tanhf(gu) + sigf(gl) * lc + sigf(gr) * rc;
        h = tanhf(c);
    }
    staging[(size_t)b * H2_ + jg]      = h;
    staging[(size_t)b * H2_ + H_ + jg] = c;
    // outputs[b, T-1-t, :] = h  (reversed along T, h part only)
    out[((size_t)b * T + (T - 1 - t)) * H_ + jg] = h;
}

// ---------------------------------------------------------------------------
// Kernel 5: commit staging -> stack[b][write_slot] (device-wide visible
// before the next step kernel by stream ordering).
// ---------------------------------------------------------------------------
__global__ __launch_bounds__(256) void commit_kernel(
    int t, const float* __restrict__ staging, const int* __restrict__ wrA,
    float* __restrict__ stack)
{
    int id = blockIdx.x * 256 + threadIdx.x;     // 0..65535
    int b   = id >> 9;
    int col = id & 511;
    int wi  = wrA[t * B_ + b];
    stack[((size_t)b * L_ + wi) * H2_ + col] = staging[(size_t)b * H2_ + col];
}

// ---------------------------------------------------------------------------
// Kernel 6: final readout h_final / c_final = stack[b][sp-1]
// ---------------------------------------------------------------------------
__global__ __launch_bounds__(256) void final_kernel(
    const float* __restrict__ stack, const int* __restrict__ spf,
    float* __restrict__ out_hc)   // out_hc = d_out + B*T*H
{
    int id = blockIdx.x * 256 + threadIdx.x;     // 0..65535
    int b   = id >> 9;
    int col = id & 511;
    float v = stack[((size_t)b * L_ + spf[b]) * H2_ + col];
    if (col < H_) out_hc[(size_t)b * H_ + col] = v;                          // h_final
    else          out_hc[(size_t)B_ * H_ + (size_t)b * H_ + (col - H_)] = v; // c_final
}

// ---------------------------------------------------------------------------
// Host launcher
// ---------------------------------------------------------------------------
extern "C" void kernel_launch(void* const* d_in, const int* in_sizes, int n_in,
                              void* d_out, int out_size, void* d_ws, size_t ws_size,
                              hipStream_t stream)
{
    const float* seq   = (const float*)d_in[0];   // (L,B,E)
    const float* Wc    = (const float*)d_in[1];   // (E,H)
    const float* bc    = (const float*)d_in[2];   // (H,)
    const float* Wo    = (const float*)d_in[3];   // (E,H)
    const float* bo    = (const float*)d_in[4];   // (H,)
    const float* Wl    = (const float*)d_in[5];   // (H,4H)
    const float* Wr    = (const float*)d_in[6];   // (H,4H)
    const float* bcomp = (const float*)d_in[7];   // (4H,)
    const int*   trans = (const int*)d_in[8];     // (T,B)
    const int T = in_sizes[8] / B_;               // 255

    // Workspace carve-up (aligned to 256B)
    char* ws = (char*)d_ws;
    size_t off = 0;
    auto carve = [&](size_t bytes) -> char* {
        char* p = ws + off;
        off += (bytes + 255) & ~(size_t)255;
        return p;
    };
    unsigned short* Bfrag   = (unsigned short*)carve((size_t)64 * 16 * 32 * 16 * 2);   // 1 MB
    unsigned short* Cfrag   = (unsigned short*)carve((size_t)16 * 8 * 32 * 16 * 2);
    unsigned short* Ofrag   = (unsigned short*)carve((size_t)16 * 8 * 32 * 16 * 2);
    unsigned short* Afrag   = (unsigned short*)carve((size_t)1024 * 8 * 32 * 16 * 2);  // 8 MB
    float* buffers          = (float*)carve((size_t)B_ * L_ * H2_ * 4);                // 32 MB
    float* stackp           = (float*)carve((size_t)B_ * L_ * H2_ * 4);                // 32 MB
    float* staging          = (float*)carve((size_t)B_ * H2_ * 4);
    int* is_shift           = (int*)carve((size_t)T * B_ * 4);
    int* liA                = (int*)carve((size_t)T * B_ * 4);
    int* riA                = (int*)carve((size_t)T * B_ * 4);
    int* biA                = (int*)carve((size_t)T * B_ * 4);
    int* wrA                = (int*)carve((size_t)T * B_ * 4);
    int* spf                = (int*)carve(512);

    float* out = (float*)d_out;

    // 1) pre-swizzle weights + activations into WMMA fragment layouts (bf16)
    prep_weights_kernel<<<2560, 256, 0, stream>>>(Wl, Wr, Wc, Wo, Bfrag, Cfrag, Ofrag);
    prep_seq_kernel<<<16384, 256, 0, stream>>>(seq, Afrag);
    // 2) precompute all stack pointer indices
    idx_kernel<<<1, 128, 0, stream>>>(trans, T, is_shift, liA, riA, biA, wrA, spf);
    // 3) leaf GEMMs + pointwise -> buffers
    leaf_kernel<<<2048, 256, 0, stream>>>(Afrag, bc, bo, Cfrag, Ofrag, buffers);
    // 4) sequential scan: WMMA step + commit per transition
    for (int t = 0; t < T; ++t) {
        step_kernel<<<128, 128, 0, stream>>>(t, T, stackp, buffers, Bfrag, bcomp,
                                             is_shift, liA, riA, biA, staging, out);
        commit_kernel<<<256, 256, 0, stream>>>(t, staging, wrA, stackp);
    }
    // 5) final h/c readout
    final_kernel<<<256, 256, 0, stream>>>(stackp, spf, out + (size_t)B_ * T * H_);
}
